// ptSDPA_39668317946373
// MI455X (gfx1250) — matrix operand order repaired and checked by reading, not compiled, not run on hardware
//
#include <hip/hip_runtime.h>
#include <math.h>

typedef __attribute__((ext_vector_type(16))) _Float16 v16h;
typedef __attribute__((ext_vector_type(8)))  _Float16 v8h;
typedef __attribute__((ext_vector_type(4)))  _Float16 v4h;
typedef __attribute__((ext_vector_type(2)))  _Float16 h2;
typedef __attribute__((ext_vector_type(8)))  float    v8f;
typedef __attribute__((ext_vector_type(8)))  int      v8i;

#define WMMA_F32_F16(a, b, c)                                           \
  __builtin_amdgcn_wmma_f32_16x16x32_f16(false, (a), false, (b),        \
                                         (short)0, (c), false, false)

#define EXP2(x) __builtin_amdgcn_exp2f(x)   // bare v_exp_f32

constexpr int HQ = 16, HKV = 4, SEQ = 2048, DIM = 64;
constexpr int KV_ELEMS = 2 * HKV * SEQ * DIM;  // 1,048,576

// ---------- prep kernel 1: K fp32 -> f16, same layout ----------
__global__ __launch_bounds__(256) void cvt_k_f16(const float* __restrict__ K,
                                                 _Float16* __restrict__ Kh) {
  const int i = (blockIdx.x * 256 + threadIdx.x);       // float4 index
  const float4 v = ((const float4*)K)[i];
  v4h o;
  o[0] = (_Float16)v.x; o[1] = (_Float16)v.y;
  o[2] = (_Float16)v.z; o[3] = (_Float16)v.w;
  ((v4h*)Kh)[i] = o;
}

// ---------- prep kernel 2: V fp32 [bh][s][d] -> f16 V^T [bh][d][s] ----------
__global__ __launch_bounds__(256) void transpose_v_f16(
    const float* __restrict__ V, _Float16* __restrict__ Vt) {
  __shared__ _Float16 tile[64][65];                     // pad: no bank conflicts
  const int bh = blockIdx.x;                            // 0..7
  const int s0 = blockIdx.y * 64;                       // 64-key slab
  const float* src = V + ((size_t)bh * SEQ + s0) * DIM;
#pragma unroll
  for (int i = threadIdx.x; i < 64 * 64; i += 256) {
    tile[i >> 6][i & 63] = (_Float16)src[i];            // coalesced read
  }
  __syncthreads();
  _Float16* dst = Vt + (size_t)bh * DIM * SEQ + s0;
#pragma unroll
  for (int i = threadIdx.x; i < 64 * 64; i += 256) {
    const int d = i >> 6, sr = i & 63;
    dst[(size_t)d * SEQ + sr] = tile[sr][d];            // coalesced write
  }
}

// ---------- main: flash attention, one wave32 per 32-query pair ----------
// S^T = K * Q^T  (keys on M axis, queries on lanes)
// O^T = V^T * P^T (d on M axis, queries on lanes)
// Softmax runs in the exp2 domain: Q is pre-scaled by 0.125*log2(e); all
// exponentials are bare v_exp_f32.
__global__ __launch_bounds__(128) void sdpa_fa_wmma(
    const float* __restrict__ Q, const _Float16* __restrict__ Kh,
    const _Float16* __restrict__ Vt, float* __restrict__ O) {
  constexpr float QSCALE = 0.125f * 1.4426950408889634f;  // 1/sqrt(64)*log2(e)
  const int lane = threadIdx.x & 31;
  const int wave = blockIdx.x * 4 + (threadIdx.x >> 5);   // 0..2047
  constexpr int QB = SEQ / 32;                            // 64 query-pair blocks
  const int b   = wave / (HQ * QB);
  const int rem = wave % (HQ * QB);
  const int h   = rem / QB;
  const int q0  = (rem % QB) * 32;
  const int hkv = h / (HQ / HKV);
  const size_t bh = (size_t)b * HKV + hkv;

  const float*    qb_   = Q  + (((size_t)b * HQ + h) * SEQ + q0) * DIM;
  const _Float16* kbase = Kh + bh * SEQ * DIM;
  const _Float16* vbase = Vt + bh * DIM * SEQ;
  float*          obase = O  + (((size_t)b * HQ + h) * SEQ + q0) * DIM;

  const int col = lane & 15, half = lane >> 4;

  // Q^T B-fragments (loop invariant, pre-scaled):
  //   lane N=col, K(e)=half*16+e -> d=dc*32+half*16+e
  v16h qf[2][2];
#pragma unroll
  for (int qi = 0; qi < 2; ++qi) {
    const float* qrow = qb_ + (size_t)(qi * 16 + col) * DIM;
#pragma unroll
    for (int dc = 0; dc < 2; ++dc) {
      const float4* qv = (const float4*)(qrow + dc * 32 + half * 16);
#pragma unroll
      for (int g = 0; g < 4; ++g) {
        const float4 v = qv[g];
        qf[qi][dc][g * 4 + 0] = (_Float16)(v.x * QSCALE);
        qf[qi][dc][g * 4 + 1] = (_Float16)(v.y * QSCALE);
        qf[qi][dc][g * 4 + 2] = (_Float16)(v.z * QSCALE);
        qf[qi][dc][g * 4 + 3] = (_Float16)(v.w * QSCALE);
      }
    }
  }

  v8f acc[2][4];
#pragma unroll
  for (int qi = 0; qi < 2; ++qi)
#pragma unroll
    for (int t = 0; t < 4; ++t) acc[qi][t] = (v8f){0, 0, 0, 0, 0, 0, 0, 0};
  float m[2] = {-INFINITY, -INFINITY};   // running max, log2 domain
  float l[2] = {0.0f, 0.0f};

  for (int kb0 = 0; kb0 < SEQ; kb0 += 32) {
    // ---- K A-fragments (two 16B loads each), shared by both q-tiles.
    //      A 16x32: M = key = kb0+tt*16+col, K(e) = half*16 + e
    v16h kf[2][2];
#pragma unroll
    for (int tt = 0; tt < 2; ++tt) {
      const _Float16* kp = kbase + (size_t)(kb0 + tt * 16 + col) * DIM;
#pragma unroll
      for (int dc = 0; dc < 2; ++dc) {
        const v8h lo = *(const v8h*)(kp + dc * 32 + half * 16);
        const v8h hi = *(const v8h*)(kp + dc * 32 + half * 16 + 8);
        kf[tt][dc] = __builtin_shufflevector(lo, hi, 0, 1, 2, 3, 4, 5, 6, 7, 8,
                                             9, 10, 11, 12, 13, 14, 15);
      }
    }
    // ---- S^T = K * Q^T : 8 WMMA (scores arrive pre-scaled by QSCALE)
    v8f s[2][2];
#pragma unroll
    for (int qi = 0; qi < 2; ++qi)
#pragma unroll
      for (int tt = 0; tt < 2; ++tt) {
        s[qi][tt] = (v8f){0, 0, 0, 0, 0, 0, 0, 0};
#pragma unroll
        for (int dc = 0; dc < 2; ++dc)
          s[qi][tt] = WMMA_F32_F16(kf[tt][dc], qf[qi][dc], s[qi][tt]);
      }

    // ---- Online softmax (exp2 domain) + packed-f16 P^T transpose.
    v16h pf[2];
#pragma unroll
    for (int qi = 0; qi < 2; ++qi) {
      float mloc = -INFINITY;
#pragma unroll
      for (int tt = 0; tt < 2; ++tt)
#pragma unroll
        for (int r = 0; r < 8; ++r) mloc = __builtin_fmaxf(mloc, s[qi][tt][r]);
      mloc = __builtin_fmaxf(mloc, __shfl_xor(mloc, 16, 32));
      const float mnew  = __builtin_fmaxf(m[qi], mloc);
      const float alpha = EXP2(m[qi] - mnew);
      m[qi] = mnew;

      float pv[2][8];
      float lsum = 0.0f;
#pragma unroll
      for (int tt = 0; tt < 2; ++tt)
#pragma unroll
        for (int r = 0; r < 8; ++r) {
          const float e = EXP2(s[qi][tt][r] - mnew);
          pv[tt][r] = e;
          lsum += e;
        }
      lsum += __shfl_xor(lsum, 16, 32);
      l[qi] = l[qi] * alpha + lsum;
#pragma unroll
      for (int t = 0; t < 4; ++t)
#pragma unroll
        for (int r = 0; r < 8; ++r) acc[qi][t][r] *= alpha;

      // pack p into f16 pairs, shuffle dwords (2 elems per ds_bpermute).
      // dest dword j of pf: elements 2j,2j+1 -> keys kb0+half*16+2j(+1):
      //   src tile = half, src VGPRs r=2j,2j+1, src lane = col (j<4) /
      //   col+16 (j>=4)
      int pk0[4], pk1[4];
#pragma unroll
      for (int j = 0; j < 4; ++j) {
        const h2 a = {(_Float16)pv[0][2 * j], (_Float16)pv[0][2 * j + 1]};
        const h2 c = {(_Float16)pv[1][2 * j], (_Float16)pv[1][2 * j + 1]};
        pk0[j] = __builtin_bit_cast(int, a);
        pk1[j] = __builtin_bit_cast(int, c);
      }
      v8i pd;
#pragma unroll
      for (int j = 0; j < 4; ++j) {
        const int s0 = __shfl(pk0[j], col, 32);        // keys half*16 + 2j..
        const int s1 = __shfl(pk1[j], col, 32);
        const int t0 = __shfl(pk0[j], col + 16, 32);   // keys half*16+8+2j..
        const int t1 = __shfl(pk1[j], col + 16, 32);
        pd[j]     = half ? s1 : s0;
        pd[j + 4] = half ? t1 : t0;
      }
      pf[qi] = __builtin_bit_cast(v16h, pd);
    }

    // ---- O^T += V^T * P^T : V^T frags are contiguous 16B loads from Vt.
#pragma unroll
    for (int t = 0; t < 4; ++t) {
      const _Float16* vrow = vbase + (size_t)(t * 16 + col) * SEQ;
      const v8h lo = *(const v8h*)(vrow + kb0 + half * 16);
      const v8h hi = *(const v8h*)(vrow + kb0 + half * 16 + 8);
      const v16h a = __builtin_shufflevector(lo, hi, 0, 1, 2, 3, 4, 5, 6, 7, 8,
                                             9, 10, 11, 12, 13, 14, 15);
#pragma unroll
      for (int qi = 0; qi < 2; ++qi)
        acc[qi][t] = WMMA_F32_F16(a, pf[qi], acc[qi][t]);
    }
  }

  // ---- Normalize + store: d = t*16 + r + 8*half contiguous in r -> float4 x2
#pragma unroll
  for (int qi = 0; qi < 2; ++qi) {
    const float invl = 1.0f / l[qi];
    float* orow = obase + (size_t)(qi * 16 + col) * DIM;
#pragma unroll
    for (int t = 0; t < 4; ++t) {
      float4 o0, o1;
      o0.x = acc[qi][t][0] * invl; o0.y = acc[qi][t][1] * invl;
      o0.z = acc[qi][t][2] * invl; o0.w = acc[qi][t][3] * invl;
      o1.x = acc[qi][t][4] * invl; o1.y = acc[qi][t][5] * invl;
      o1.z = acc[qi][t][6] * invl; o1.w = acc[qi][t][7] * invl;
      float* dp = orow + t * 16 + 8 * half;
      ((float4*)dp)[0] = o0;
      ((float4*)dp)[1] = o1;
    }
  }
}

extern "C" void kernel_launch(void* const* d_in, const int* in_sizes, int n_in,
                              void* d_out, int out_size, void* d_ws,
                              size_t ws_size, hipStream_t stream) {
  (void)in_sizes; (void)n_in; (void)out_size; (void)ws_size;
  const float* q = (const float*)d_in[0];
  const float* k = (const float*)d_in[1];
  const float* v = (const float*)d_in[2];
  float* o = (float*)d_out;
  _Float16* kh = (_Float16*)d_ws;                 // 2 MB
  _Float16* vt = kh + (size_t)KV_ELEMS;           // 2 MB

  hipLaunchKernelGGL(cvt_k_f16, dim3(KV_ELEMS / 4 / 256), dim3(256), 0, stream,
                     k, kh);
  hipLaunchKernelGGL(transpose_v_f16, dim3(2 * HKV, SEQ / 64), dim3(256), 0,
                     stream, v, vt);
  // 2*16*(2048/32) = 2048 waves, 4 waves per block
  hipLaunchKernelGGL(sdpa_fa_wmma, dim3(512), dim3(128), 0, stream, q, kh, vt,
                     o);
}
